// Attention_84129819394116
// MI455X (gfx1250) — matrix operand order repaired and checked by reading, not compiled, not run on hardware
//
#include <hip/hip_runtime.h>
#include <math.h>

typedef __attribute__((ext_vector_type(16))) __bf16 v16bf;
typedef __attribute__((ext_vector_type(8)))  float  v8f;
typedef __attribute__((ext_vector_type(4)))  int    v4i;

#define DIM   1024
#define NH    16
#define HD    64
#define BATCH 2
#define SEQ   2048
#define ROWS  (BATCH * SEQ)   // 4096

// ---------------------------------------------------------------------------
// Async global->LDS staging (CDNA5 ASYNCcnt path) with fallback
// ---------------------------------------------------------------------------
#if __has_builtin(__builtin_amdgcn_global_load_async_to_lds_b128) && \
    __has_builtin(__builtin_amdgcn_s_wait_asynccnt)
#define USE_ASYNC_LDS 1
#endif

#ifdef USE_ASYNC_LDS
#define AS1 __attribute__((address_space(1)))
#define AS3 __attribute__((address_space(3)))
__device__ __forceinline__ void stage16(__bf16* lds, const __bf16* g) {
  __builtin_amdgcn_global_load_async_to_lds_b128((AS1 v4i*)g, (AS3 v4i*)lds,
                                                 0, 0);
}
__device__ __forceinline__ void stage_wait() {
  __builtin_amdgcn_s_wait_asynccnt(0);
  __syncthreads();
}
#else
__device__ __forceinline__ void stage16(__bf16* lds, const __bf16* g) {
  *(uint4*)lds = *(const uint4*)g;
}
__device__ __forceinline__ void stage_wait() { __syncthreads(); }
#endif

// Assemble a 16xbf16 fragment from two contiguous 16-byte runs (-> 2x b128)
__device__ __forceinline__ v16bf frag2(const __bf16* p0, const __bf16* p1) {
  union { v16bf v; uint4 q[2]; } u;
  u.q[0] = *(const uint4*)p0;
  u.q[1] = *(const uint4*)p1;
  return u.v;
}

// ---------------------------------------------------------------------------
// fp32 -> bf16 (row-major, vectorized: 8 elems/thread)
// ---------------------------------------------------------------------------
__global__ __launch_bounds__(256) void cvt_f32_bf16(const float* __restrict__ in,
                                                    __bf16* __restrict__ out) {
  const int i = (blockIdx.x * 256 + threadIdx.x) * 8;
  float4 f0 = *(const float4*)(in + i);
  float4 f1 = *(const float4*)(in + i + 4);
  __bf16 t[8] = {(__bf16)f0.x, (__bf16)f0.y, (__bf16)f0.z, (__bf16)f0.w,
                 (__bf16)f1.x, (__bf16)f1.y, (__bf16)f1.z, (__bf16)f1.w};
  *(uint4*)(out + i) = *(uint4*)t;
}

// ---------------------------------------------------------------------------
// fp32 [k][n] -> bf16 transposed [n][k]  (weights: makes B-frags contiguous)
// ---------------------------------------------------------------------------
__global__ __launch_bounds__(256) void cvt_w_transpose(const float* __restrict__ w,
                                                       __bf16* __restrict__ wt) {
  __shared__ float tile[32][33];
  const int bn = blockIdx.x * 32;       // n block
  const int bk = blockIdx.y * 32;       // k block
  const int tx = threadIdx.x & 31, ty = threadIdx.x >> 5;  // ty: 0..7
#pragma unroll
  for (int i = 0; i < 32; i += 8)
    tile[ty + i][tx] = w[(size_t)(bk + ty + i) * DIM + bn + tx];
  __syncthreads();
#pragma unroll
  for (int i = 0; i < 32; i += 8)
    wt[(size_t)(bn + ty + i) * DIM + bk + tx] = (__bf16)tile[tx][ty + i];
}

// ---------------------------------------------------------------------------
// Tiled bf16 GEMM: C[M,N](f32) = A[M,K](bf16, row-major) * Bt[N,K](bf16)^T
// Block tile 128x64, K-step 64; 8 waves, each owns a 16-row strip x 64 cols.
// All fragments preloaded per staging round, then 8 WMMAs issue back-to-back.
// ---------------------------------------------------------------------------
#define GTM 128
#define GTN 64
#define GTK 64
#define LDA_S 72   // elements; 144B rows (16B-aligned runs)
#define LDB_S 72

__global__ __launch_bounds__(256) void gemm_bf16_wmma(
    const __bf16* __restrict__ A, const __bf16* __restrict__ Bt,
    float* __restrict__ C, int M, int N, int K) {
  __shared__ __bf16 As[GTM * LDA_S];   // [m][k]
  __shared__ __bf16 Bs[GTN * LDB_S];   // [n][k]
  const int tid  = threadIdx.x;
  const int lane = tid & 31;
  const int wave = tid >> 5;           // m-strip 0..7
  const int half = lane >> 4;          // 0/1
  const int l15  = lane & 15;
  const int m0 = blockIdx.y * GTM;
  const int n0 = blockIdx.x * GTN;

  v8f acc[4];
  v8f z = {};
#pragma unroll
  for (int t = 0; t < 4; ++t) acc[t] = z;

  for (int k0 = 0; k0 < K; k0 += GTK) {
    // stage A tile: 128x64 -> 4 x async b128 per thread
#pragma unroll
    for (int i = 0; i < 4; ++i) {
      const int chunk = tid * 4 + i;           // 0..1023
      const int r = chunk >> 3, c = (chunk & 7) * 8;
      stage16(&As[r * LDA_S + c], A + (size_t)(m0 + r) * K + k0 + c);
    }
    // stage Bt tile: 64x64 -> 2 x async b128 per thread
#pragma unroll
    for (int i = 0; i < 2; ++i) {
      const int chunk = tid * 2 + i;           // 0..511
      const int r = chunk >> 3, c = (chunk & 7) * 8;
      stage16(&Bs[r * LDB_S + c], Bt + (size_t)(n0 + r) * K + k0 + c);
    }
    if (k0 + GTK < K) {
      __builtin_prefetch(A + (size_t)(m0 + (tid >> 1)) * K + k0 + GTK, 0, 1);
      __builtin_prefetch(Bt + (size_t)(n0 + (tid >> 2)) * K + k0 + GTK, 0, 1);
    }
    stage_wait();

    // preload ALL fragments for this round (ds_load_b128 pairs) ...
    v16bf a[2], b[2][4];
#pragma unroll
    for (int ks = 0; ks < 2; ++ks) {
      const __bf16* arow = &As[(wave * 16 + l15) * LDA_S + ks * 32 + half * 8];
      a[ks] = frag2(arow, arow + 16);
#pragma unroll
      for (int t = 0; t < 4; ++t) {
        const __bf16* brow = &Bs[(t * 16 + l15) * LDB_S + ks * 32 + half * 8];
        b[ks][t] = frag2(brow, brow + 16);
      }
    }
    // ... then issue 8 WMMAs back-to-back (XDL pipe stays fed)
#pragma unroll
    for (int ks = 0; ks < 2; ++ks)
#pragma unroll
      for (int t = 0; t < 4; ++t)
        acc[t] = __builtin_amdgcn_wmma_f32_16x16x32_bf16(false, a[ks], false,
                                                         b[ks][t], (short)0,
                                                         acc[t], false, false);
    __syncthreads();
  }

  const int rbase = m0 + wave * 16 + half * 8;
#pragma unroll
  for (int t = 0; t < 4; ++t) {
    const int col = n0 + t * 16 + l15;
#pragma unroll
    for (int r = 0; r < 8; ++r)
      C[(size_t)(rbase + r) * N + col] = acc[t][r];
  }
}

// ---------------------------------------------------------------------------
// LayerNorm + RoPE + bf16, permuting [b,s,h*64+d] -> [(b,h),s,d]
// ---------------------------------------------------------------------------
__global__ __launch_bounds__(256) void ln_rope_bf16(
    const float* __restrict__ qf, const float* __restrict__ kf,
    const float* __restrict__ q_scale, const float* __restrict__ q_bias,
    const float* __restrict__ k_scale, const float* __restrict__ k_bias,
    const float* __restrict__ fsin, const float* __restrict__ fcos,
    __bf16* __restrict__ qb, __bf16* __restrict__ kb) {
  const int row  = blockIdx.x;                // b*SEQ + s
  const bool isk = blockIdx.y != 0;
  const float* src   = (isk ? kf : qf) + (size_t)row * DIM;
  const float* scale = isk ? k_scale : q_scale;
  const float* bias  = isk ? k_bias  : q_bias;
  __bf16* dst = isk ? kb : qb;
  const int s  = row & (SEQ - 1);
  const int bq = row >> 11;
  const int tid = threadIdx.x;

  float v[4];
  float sum = 0.f, sq = 0.f;
#pragma unroll
  for (int j = 0; j < 4; ++j) {
    float x = src[tid * 4 + j];
    v[j] = x; sum += x; sq += x * x;
  }
#pragma unroll
  for (int off = 16; off > 0; off >>= 1) {
    sum += __shfl_xor(sum, off, 32);
    sq  += __shfl_xor(sq,  off, 32);
  }
  __shared__ float s_sum[8], s_sq[8], s_stats[2];
  if ((tid & 31) == 0) { s_sum[tid >> 5] = sum; s_sq[tid >> 5] = sq; }
  __syncthreads();
  if (tid == 0) {
    float ts = 0.f, tq = 0.f;
    for (int w = 0; w < 8; ++w) { ts += s_sum[w]; tq += s_sq[w]; }
    float mu  = ts / (float)DIM;
    float var = tq / (float)DIM - mu * mu;
    s_stats[0] = mu;
    s_stats[1] = rsqrtf(var + 1e-6f);
  }
  __syncthreads();
  const float mu = s_stats[0], rstd = s_stats[1];

#pragma unroll
  for (int p = 0; p < 2; ++p) {
    const int c0 = tid * 4 + p * 2;
    float re = (v[p * 2]     - mu) * rstd * scale[c0]     + bias[c0];
    float im = (v[p * 2 + 1] - mu) * rstd * scale[c0 + 1] + bias[c0 + 1];
    const int h  = c0 >> 6;
    const int d  = c0 & (HD - 1);
    const int jj = d >> 1;
    const float cs = fcos[s * (HD / 2) + jj];
    const float sn = fsin[s * (HD / 2) + jj];
    const size_t base = (((size_t)bq * NH + h) * SEQ + s) * HD + d;
    dst[base]     = (__bf16)(re * cs - im * sn);
    dst[base + 1] = (__bf16)(re * sn + im * cs);
  }
}

// ---------------------------------------------------------------------------
// V: fp32 [b,s,h*64+d] -> bf16 TRANSPOSED [(b,h), d, s]
// ---------------------------------------------------------------------------
__global__ __launch_bounds__(256) void v_perm_bf16(const float* __restrict__ vf,
                                                   __bf16* __restrict__ vt) {
  const int row = blockIdx.x;
  const int s  = row & (SEQ - 1);
  const int bq = row >> 11;
  const int tid = threadIdx.x;
#pragma unroll
  for (int j = 0; j < 4; ++j) {
    const int c = tid * 4 + j;
    const int h = c >> 6, d = c & (HD - 1);
    vt[(((size_t)bq * NH + h) * HD + d) * SEQ + s] = (__bf16)vf[(size_t)row * DIM + c];
  }
}

// ---------------------------------------------------------------------------
// Flash-style attention. Q:[bh][s][d], K:[bh][s][d], Vt:[bh][d][s] (bf16)
// 8 waves x 16-row Q strips; 32-key blocks; frags preloaded, WMMAs batched.
// ---------------------------------------------------------------------------
#define KB   32
#define LDK  72   // Ks rows [key][d]: 144B
#define LDV  40   // Vts rows [d][key]: 80B
#define LDP  40   // Ps rows [qrow][key]: 80B

__global__ __launch_bounds__(256) void attn_wmma(
    const __bf16* __restrict__ Q, const __bf16* __restrict__ Kx,
    const __bf16* __restrict__ Vt, __bf16* __restrict__ Out) {
  __shared__ __bf16 Ks[KB * LDK];          // [key][d]
  __shared__ __bf16 Vts[HD * LDV];         // [d][key]
  __shared__ __bf16 Ps[8][16 * LDP];       // per-wave 16x32 probs

  const int bh = blockIdx.y;
  const __bf16* q  = Q  + (size_t)bh * SEQ * HD;
  const __bf16* k  = Kx + (size_t)bh * SEQ * HD;
  const __bf16* vt = Vt + (size_t)bh * HD * SEQ;

  const int tid  = threadIdx.x;
  const int lane = tid & 31;
  const int wave = tid >> 5;
  const int half = lane >> 4;
  const int l15  = lane & 15;
  const int q0   = blockIdx.x * 128 + wave * 16;

  // Q strip (16x64) as two A-frags: 2x b128 global loads each
  v16bf aq[2];
#pragma unroll
  for (int ks = 0; ks < 2; ++ks) {
    const __bf16* qrow = q + (size_t)(q0 + l15) * HD + ks * 32 + half * 8;
    aq[ks] = frag2(qrow, qrow + 16);
  }

  v8f o[4];
  v8f zero = {};
#pragma unroll
  for (int t = 0; t < 4; ++t) o[t] = zero;
  float m_run[8], l_run[8];
#pragma unroll
  for (int r = 0; r < 8; ++r) { m_run[r] = -1e30f; l_run[r] = 0.f; }

  const float sc = 0.125f;  // 1/sqrt(64)

  for (int kb = 0; kb < SEQ; kb += KB) {
    __syncthreads();
    {  // stage K block 32x64 and V^T block 64x32: 1 x async b128 per thread
      const int rk = tid >> 3, ck = (tid & 7) * 8;
      stage16(&Ks[rk * LDK + ck], k + (size_t)(kb + rk) * HD + ck);
      const int rv = tid >> 2, cv = (tid & 3) * 8;
      stage16(&Vts[rv * LDV + cv], vt + (size_t)rv * SEQ + kb + cv);
    }
    stage_wait();

    // preload K and V fragments for this block
    v16bf bk[2][2], bv[4];
#pragma unroll
    for (int nsub = 0; nsub < 2; ++nsub)
#pragma unroll
      for (int ks = 0; ks < 2; ++ks) {
        const __bf16* krow = &Ks[(nsub * 16 + l15) * LDK + ks * 32 + half * 8];
        bk[nsub][ks] = frag2(krow, krow + 16);
      }
#pragma unroll
    for (int t = 0; t < 4; ++t) {
      const __bf16* vrow = &Vts[(t * 16 + l15) * LDV + half * 8];
      bv[t] = frag2(vrow, vrow + 16);
    }

    // scores 16x32: two 16x16 tiles, K-dim 64; 4 WMMAs back-to-back
    v8f sfrag[2];
#pragma unroll
    for (int nsub = 0; nsub < 2; ++nsub) {
      v8f c0 = zero;
#pragma unroll
      for (int ks = 0; ks < 2; ++ks)
        c0 = __builtin_amdgcn_wmma_f32_16x16x32_bf16(false, aq[ks], false,
                                                     bk[nsub][ks], (short)0,
                                                     c0, false, false);
      sfrag[nsub] = c0;
    }

    // online softmax (rows: r + 8*half across wave16 groups)
    float alpha[8];
#pragma unroll
    for (int r = 0; r < 8; ++r) {
      float mx = fmaxf(sfrag[0][r] * sc, sfrag[1][r] * sc);
#pragma unroll
      for (int off = 8; off > 0; off >>= 1)
        mx = fmaxf(mx, __shfl_xor(mx, off, 16));
      const float mo = m_run[r];
      const float mn = fmaxf(mo, mx);
      alpha[r] = __expf(mo - mn);
      m_run[r] = mn;
    }
    float psum[8];
#pragma unroll
    for (int r = 0; r < 8; ++r) psum[r] = 0.f;
#pragma unroll
    for (int nsub = 0; nsub < 2; ++nsub)
#pragma unroll
      for (int r = 0; r < 8; ++r) {
        float p = __expf(sfrag[nsub][r] * sc - m_run[r]);
        sfrag[nsub][r] = p;
        psum[r] += p;
      }
#pragma unroll
    for (int r = 0; r < 8; ++r) {
      float ps = psum[r];
#pragma unroll
      for (int off = 8; off > 0; off >>= 1)
        ps += __shfl_xor(ps, off, 16);
      l_run[r] = l_run[r] * alpha[r] + ps;
    }
#pragma unroll
    for (int t = 0; t < 4; ++t)
#pragma unroll
      for (int r = 0; r < 8; ++r)
        o[t][r] *= alpha[r];

    // P: C-layout -> LDS (scatter) -> A-layout (2x b128); per-wave, in-order
    __bf16* ps_ = Ps[wave];
#pragma unroll
    for (int nsub = 0; nsub < 2; ++nsub)
#pragma unroll
      for (int r = 0; r < 8; ++r)
        ps_[(r + half * 8) * LDP + nsub * 16 + l15] = (__bf16)sfrag[nsub][r];
    const __bf16* prow = &ps_[l15 * LDP + half * 8];
    const v16bf ap = frag2(prow, prow + 16);

    // O += P x V: 4 WMMAs back-to-back
#pragma unroll
    for (int t = 0; t < 4; ++t)
      o[t] = __builtin_amdgcn_wmma_f32_16x16x32_bf16(false, ap, false, bv[t],
                                                     (short)0, o[t], false, false);
  }

  // epilogue: normalize, write [b, s, h*64+d] bf16
  const int b_ = bh / NH, h_ = bh % NH;
#pragma unroll
  for (int t = 0; t < 4; ++t) {
    const int col = h_ * HD + t * 16 + l15;
#pragma unroll
    for (int r = 0; r < 8; ++r) {
      const int row = q0 + r + half * 8;
      Out[((size_t)b_ * SEQ + row) * DIM + col] = (__bf16)(o[t][r] / l_run[r]);
    }
  }
}

// ---------------------------------------------------------------------------
extern "C" void kernel_launch(void* const* d_in, const int* in_sizes, int n_in,
                              void* d_out, int out_size, void* d_ws, size_t ws_size,
                              hipStream_t stream) {
  const float* x       = (const float*)d_in[0];
  const float* fsin    = (const float*)d_in[1];
  const float* fcos    = (const float*)d_in[2];
  const float* wq      = (const float*)d_in[3];
  const float* wk      = (const float*)d_in[4];
  const float* wv      = (const float*)d_in[5];
  const float* wo      = (const float*)d_in[6];
  const float* q_scale = (const float*)d_in[7];
  const float* q_bias  = (const float*)d_in[8];
  const float* k_scale = (const float*)d_in[9];
  const float* k_bias  = (const float*)d_in[10];
  float* out = (float*)d_out;

  char* ws = (char*)d_ws;
  size_t off = 0;
  auto carve = [&](size_t bytes) -> void* {
    void* p = ws + off;
    off += (bytes + 255) & ~(size_t)255;
    return p;
  };
  __bf16* xb  = (__bf16*)carve((size_t)ROWS * DIM * 2);
  __bf16* wqt = (__bf16*)carve((size_t)DIM * DIM * 2);
  __bf16* wkt = (__bf16*)carve((size_t)DIM * DIM * 2);
  __bf16* wvt = (__bf16*)carve((size_t)DIM * DIM * 2);
  __bf16* wot = (__bf16*)carve((size_t)DIM * DIM * 2);
  float*  qf  = (float*)carve((size_t)ROWS * DIM * 4);
  float*  kf  = (float*)carve((size_t)ROWS * DIM * 4);
  float*  vf  = (float*)carve((size_t)ROWS * DIM * 4);
  __bf16* qb  = (__bf16*)carve((size_t)ROWS * DIM * 2);
  __bf16* kb  = (__bf16*)carve((size_t)ROWS * DIM * 2);
  __bf16* vtb = (__bf16*)carve((size_t)ROWS * DIM * 2);
  __bf16* ab  = (__bf16*)carve((size_t)ROWS * DIM * 2);

  cvt_f32_bf16<<<ROWS * DIM / 2048, 256, 0, stream>>>(x, xb);
  dim3 tg(32, 32);
  cvt_w_transpose<<<tg, 256, 0, stream>>>(wq, wqt);
  cvt_w_transpose<<<tg, 256, 0, stream>>>(wk, wkt);
  cvt_w_transpose<<<tg, 256, 0, stream>>>(wv, wvt);
  cvt_w_transpose<<<tg, 256, 0, stream>>>(wo, wot);

  dim3 gg(DIM / GTN, ROWS / GTM);  // (16, 32)
  gemm_bf16_wmma<<<gg, 256, 0, stream>>>(xb, wqt, qf, ROWS, DIM, DIM);
  gemm_bf16_wmma<<<gg, 256, 0, stream>>>(xb, wkt, kf, ROWS, DIM, DIM);
  gemm_bf16_wmma<<<gg, 256, 0, stream>>>(xb, wvt, vf, ROWS, DIM, DIM);

  ln_rope_bf16<<<dim3(ROWS, 2), 256, 0, stream>>>(qf, kf, q_scale, q_bias,
                                                  k_scale, k_bias, fsin, fcos,
                                                  qb, kb);
  v_perm_bf16<<<ROWS, 256, 0, stream>>>(vf, vtb);

  attn_wmma<<<dim3(SEQ / 128, BATCH * NH), 256, 0, stream>>>(qb, kb, vtb, ab);

  gemm_bf16_wmma<<<gg, 256, 0, stream>>>(ab, wot, out, ROWS, DIM, DIM);
}
